// TorchCosineLoss_51376398794762
// MI455X (gfx1250) — matrix operand order repaired and checked by reading, not compiled, not run on hardware
//
#include <hip/hip_runtime.h>
#include <hip/hip_bf16.h>
#include <math.h>

// Problem constants (from reference)
#define BT_N   2048
#define H_T    2048
#define H_S    1024
#define V_N    32000
// Tiling
#define ROWS   64
#define COLS   256
#define KC     64               // bf16 K-chunk staged per iteration (2 WMMA k-steps)
#define NRB    (BT_N / ROWS)    // 32 row blocks
#define NCB    (V_N / COLS)     // 125 col blocks
#define LDAB   72               // ushort row stride for staged bf16 tiles (64 + 8 pad)
#define LDOUT  (COLS + 4)       // float row stride for logits tiles in LDS

typedef __attribute__((ext_vector_type(16))) __bf16       v16bf;
typedef __attribute__((ext_vector_type(8)))  float        v8f;
typedef __attribute__((ext_vector_type(4)))  unsigned int v4u;
typedef __attribute__((ext_vector_type(4)))  int          v4i;

// address-space qualified element typedefs for the async-copy builtin
typedef __attribute__((address_space(1))) v4i g_v4i;   // global
typedef __attribute__((address_space(3))) v4i l_v4i;   // LDS

union FragBF { v16bf v; v4u q[2]; };

#if defined(__HIP_DEVICE_COMPILE__) && __has_builtin(__builtin_amdgcn_global_load_async_to_lds_b128)
#define HAVE_ASYNC_LDS 1
#else
#define HAVE_ASYNC_LDS 0
#endif

__device__ __forceinline__ unsigned short f2bf(float f) {
    unsigned int x = __float_as_uint(f);
    unsigned int r = (x + 0x7FFFu + ((x >> 16) & 1u)) >> 16;
    return (unsigned short)r;
}

// 16-byte copy global -> LDS. Async (CDNA5 GLOBAL_LOAD_ASYNC_TO_LDS_B128) when
// the builtin exists, else plain load+store.
__device__ __forceinline__ void copy16_g2l(const unsigned short* g, unsigned short* l) {
#if HAVE_ASYNC_LDS
    g_v4i* gp = (g_v4i*)(uintptr_t)g;                       // generic -> global AS via int
    l_v4i* lp = (l_v4i*)(unsigned int)(uintptr_t)l;         // low 32 bits == LDS offset
    __builtin_amdgcn_global_load_async_to_lds_b128(gp, lp, 0, 0);
#else
    *(v4u*)l = *(const v4u*)g;
#endif
}

__device__ __forceinline__ void wait_g2l() {
#if HAVE_ASYNC_LDS
#if __has_builtin(__builtin_amdgcn_s_wait_asynccnt)
    __builtin_amdgcn_s_wait_asynccnt(0);
#else
    asm volatile("s_wait_asynccnt 0" ::: "memory");
#endif
#endif
}

// -------- one-shot f32 -> bf16 conversion (round-to-nearest-even) --------
__global__ void __launch_bounds__(256) cvt_bf16_kernel(
    const float* __restrict__ src, unsigned short* __restrict__ dst, int n4)
{
    const int i = blockIdx.x * blockDim.x + threadIdx.x;
    if (i >= n4) return;
    const float4 f = ((const float4*)src)[i];
    uint2 p;
    p.x = (unsigned int)f2bf(f.x) | ((unsigned int)f2bf(f.y) << 16);
    p.y = (unsigned int)f2bf(f.z) | ((unsigned int)f2bf(f.w) << 16);
    ((uint2*)dst)[i] = p;
}

// Compute a ROWSxCOLS logits tile: act[rowbase.., K] * wgt[colbase.., K]^T + bias
// (bf16 operands, f32 accumulate). Result -> sOut (float, LDOUT stride).
__device__ __forceinline__ void gemm_tile_bf16(
    const unsigned short* __restrict__ act, int K,
    const unsigned short* __restrict__ wgt,
    const float* __restrict__ bias,
    unsigned short* sA, unsigned short* sB, float* sOut,
    int rowbase, int colbase, int tid)
{
    const int w    = tid >> 5;          // wave id 0..7
    const int lane = tid & 31;
    const int rsub = w & 3;             // 16-row subtile within 64 rows
    const int cgrp = w >> 2;            // 0..1 -> 8 col subtiles each
    const int m    = lane & 15;
    const int hi   = lane >> 4;         // lane half select

    v8f acc[8] = {};                    // 8 C tiles of 16x16 per wave

#define LOAD_FB(dst, ksv, jv)                                                         \
    do {                                                                              \
        const unsigned short* bp =                                                    \
            sB + ((cgrp * 8 + (jv)) * 16 + m) * LDAB + (ksv) * 32 + hi * 8;           \
        (dst).q[0] = *(const v4u*)bp;                                                 \
        (dst).q[1] = *(const v4u*)(bp + 16);                                          \
    } while (0)

    const int nk = K / KC;
    for (int kk = 0; kk < nk; ++kk) {
        const int k0 = kk * KC;
        // stage A: 64 x 64 bf16 = 512 x 16B chunks (2 per thread)
        #pragma unroll
        for (int i = 0; i < 2; ++i) {
            const int ch = tid + i * 256;
            const int r = ch >> 3, c = ch & 7;     // c in 8-ushort units
            copy16_g2l(act + (size_t)(rowbase + r) * K + k0 + c * 8,
                       sA + r * LDAB + c * 8);
        }
        // stage B: 256 x 64 bf16 = 2048 x 16B chunks (8 per thread)
        #pragma unroll
        for (int i = 0; i < 8; ++i) {
            const int ch = tid + i * 256;
            const int r = ch >> 3, c = ch & 7;
            copy16_g2l(wgt + (size_t)(colbase + r) * K + k0 + c * 8,
                       sB + r * LDAB + c * 8);
        }
        wait_g2l();
        __syncthreads();

        // preload both A fragments (ks = 0,1):
        // lanes 0-15 rows M=lane K {0..7,16..23}; lanes 16-31 K {8..15,24..31}
        FragBF fa[2];
        const unsigned short* ap = sA + (rsub * 16 + m) * LDAB + hi * 8;
        fa[0].q[0] = *(const v4u*)ap;
        fa[0].q[1] = *(const v4u*)(ap + 16);
        fa[1].q[0] = *(const v4u*)(ap + 32);
        fa[1].q[1] = *(const v4u*)(ap + 48);

        #pragma unroll
        for (int ks = 0; ks < 2; ++ks) {           // two 16x16x32 k-steps
            // double-buffered B fragments: load j+1 while WMMA consumes j
            FragBF fb[2];
            LOAD_FB(fb[0], ks, 0);
            #pragma unroll
            for (int j = 0; j < 8; ++j) {
                if (j < 7) LOAD_FB(fb[(j + 1) & 1], ks, j + 1);
                acc[j] = __builtin_amdgcn_wmma_f32_16x16x32_bf16(
                    false, fa[ks].v, false, fb[j & 1].v, (short)0, acc[j], false, false);
            }
        }
        __syncthreads();
    }
#undef LOAD_FB

    // write C tiles (+bias) to sOut; C layout: VGPR v -> M = v + 8*hi, N = lane&15
    #pragma unroll
    for (int j = 0; j < 8; ++j) {
        const int csub = cgrp * 8 + j;
        const float bv = bias[colbase + csub * 16 + m];
        const int row0 = rsub * 16 + hi * 8;
        #pragma unroll
        for (int vv = 0; vv < 8; ++vv)
            sOut[(row0 + vv) * LDOUT + csub * 16 + m] = acc[j][vv] + bv;
    }
}

__global__ void __launch_bounds__(256) distill_gemm_stats_kernel(
    const unsigned short* __restrict__ si, const unsigned short* __restrict__ ti,
    const int* __restrict__ tgt,
    const unsigned short* __restrict__ wS, const float* __restrict__ bS,
    const unsigned short* __restrict__ wT, const float* __restrict__ bT,
    float* __restrict__ pm, float* __restrict__ pse, float* __restrict__ pdot,
    float* __restrict__ pss, float* __restrict__ ptt, float* __restrict__ ptgt)
{
    extern __shared__ char smem[];
    float* sS = (float*)smem;                                   // 64 x 260 f32
    float* sT = sS + ROWS * LDOUT;                              // 64 x 260 f32
    unsigned short* sA = (unsigned short*)(sT + ROWS * LDOUT);  // 64 x 72 bf16
    unsigned short* sB = sA + ROWS * LDAB;                      // 256 x 72 bf16

    const int tid     = threadIdx.x;
    const int rowbase = blockIdx.x * ROWS;
    const int cb      = blockIdx.y;
    const int colbase = cb * COLS;

    gemm_tile_bf16(si, H_S, wS, bS, sA, sB, sS, rowbase, colbase, tid);
    __syncthreads();
    gemm_tile_bf16(ti, H_T, wT, bT, sA, sB, sT, rowbase, colbase, tid);
    __syncthreads();

    // ---- per-row partial statistics over this 256-col slab ----
    const int r = tid >> 2;          // row within tile (4 threads per row)
    const int q = tid & 3;           // 64-col quarter
    const int grow = rowbase + r;
    const float* srow = sS + r * LDOUT + q * 64;
    const float* trow = sT + r * LDOUT + q * 64;

    float mx = -3.0e38f;
    #pragma unroll 8
    for (int c = 0; c < 64; ++c) mx = fmaxf(mx, srow[c]);
    mx = fmaxf(mx, __shfl_xor(mx, 1, 32));
    mx = fmaxf(mx, __shfl_xor(mx, 2, 32));   // row max across 4 threads

    const int t = tgt[grow];
    float se = 0.f, dp = 0.f, ss = 0.f, tt = 0.f;
    #pragma unroll 4
    for (int c = 0; c < 64; ++c) {
        const float sv = srow[c];
        const float tv = trow[c];
        se += __expf(sv - mx);
        dp += sv * tv;
        ss += sv * sv;
        tt += tv * tv;
        if (colbase + q * 64 + c == t) ptgt[grow] = sv;   // unique writer per row
    }
    se += __shfl_xor(se, 1, 32); se += __shfl_xor(se, 2, 32);
    dp += __shfl_xor(dp, 1, 32); dp += __shfl_xor(dp, 2, 32);
    ss += __shfl_xor(ss, 1, 32); ss += __shfl_xor(ss, 2, 32);
    tt += __shfl_xor(tt, 1, 32); tt += __shfl_xor(tt, 2, 32);

    if (q == 0) {
        const size_t o = (size_t)cb * BT_N + grow;
        pm[o] = mx; pse[o] = se; pdot[o] = dp; pss[o] = ss; ptt[o] = tt;
    }
}

__global__ void __launch_bounds__(256) row_finalize_kernel(
    const float* __restrict__ pm, const float* __restrict__ pse,
    const float* __restrict__ pdot, const float* __restrict__ pss,
    const float* __restrict__ ptt, const float* __restrict__ ptgt,
    const int* __restrict__ tgt,
    float* __restrict__ rowHard, float* __restrict__ rowSoft, float* __restrict__ rowValid)
{
    const int r = blockIdx.x * blockDim.x + threadIdx.x;
    if (r >= BT_N) return;

    float m = -3.0e38f;
    for (int cb = 0; cb < NCB; ++cb) m = fmaxf(m, pm[(size_t)cb * BT_N + r]);

    float se = 0.f, dp = 0.f, ss = 0.f, tt = 0.f;
    for (int cb = 0; cb < NCB; ++cb) {
        const size_t o = (size_t)cb * BT_N + r;
        se += pse[o] * __expf(pm[o] - m);
        dp += pdot[o]; ss += pss[o]; tt += ptt[o];
    }
    const int t = tgt[r];
    const bool valid = (t != -100);
    const float lse = m + __logf(se);
    const float nll = lse - ptgt[r];
    const float ns = fmaxf(sqrtf(ss), 1e-12f);
    const float nt = fmaxf(sqrtf(tt), 1e-12f);
    rowHard[r]  = valid ? nll : 0.f;
    rowValid[r] = valid ? 1.f : 0.f;
    rowSoft[r]  = 1.0f - dp / (ns * nt);
}

__global__ void __launch_bounds__(256) final_reduce_kernel(
    const float* __restrict__ rowHard, const float* __restrict__ rowSoft,
    const float* __restrict__ rowValid, float* __restrict__ out)
{
    __shared__ float sh[256], so[256], sv[256];
    const int tid = threadIdx.x;
    float h = 0.f, s = 0.f, v = 0.f;
    for (int r = tid; r < BT_N; r += 256) { h += rowHard[r]; s += rowSoft[r]; v += rowValid[r]; }
    sh[tid] = h; so[tid] = s; sv[tid] = v;
    __syncthreads();
    for (int off = 128; off > 0; off >>= 1) {
        if (tid < off) { sh[tid] += sh[tid + off]; so[tid] += so[tid + off]; sv[tid] += sv[tid + off]; }
        __syncthreads();
    }
    if (tid == 0) {
        const float nv = fmaxf(sv[0], 1.0f);
        out[0] = 0.5f * (sh[0] / nv) + 0.5f * (so[0] / (float)BT_N);
    }
}

extern "C" void kernel_launch(void* const* d_in, const int* in_sizes, int n_in,
                              void* d_out, int out_size, void* d_ws, size_t ws_size,
                              hipStream_t stream) {
    const float* si = (const float*)d_in[0];   // student_input [2048,1024]
    const float* ti = (const float*)d_in[1];   // teacher_input [2048,2048]
    const int*   tg = (const int*)  d_in[2];   // target [2048]
    const float* wS = (const float*)d_in[3];   // w_student [32000,1024]
    const float* bS = (const float*)d_in[4];   // b_student [32000]
    const float* wT = (const float*)d_in[5];   // w_teacher [32000,2048]
    const float* bT = (const float*)d_in[6];   // b_teacher [32000]
    float* out = (float*)d_out;

    // ---- workspace carve ----
    // bf16 mirrors (bytes): si 4MB, ti 8MB, wS 62.5MB, wT 125MB; then f32 partials (~5.2MB).
    const size_t nsi = (size_t)BT_N * H_S;     // 2,097,152
    const size_t nti = (size_t)BT_N * H_T;     // 4,194,304
    const size_t nwS = (size_t)V_N * H_S;      // 32,768,000
    const size_t nwT = (size_t)V_N * H_T;      // 65,536,000

    unsigned short* bsi = (unsigned short*)d_ws;
    unsigned short* bti = bsi + nsi;
    unsigned short* bwS = bti + nti;
    unsigned short* bwT = bwS + nwS;
    float* pm      = (float*)(bwT + nwT);          // NCB*BT each
    float* pse     = pm   + (size_t)NCB * BT_N;
    float* pdot    = pse  + (size_t)NCB * BT_N;
    float* pss     = pdot + (size_t)NCB * BT_N;
    float* ptt     = pss  + (size_t)NCB * BT_N;
    float* ptgt    = ptt  + (size_t)NCB * BT_N;    // BT
    float* rowHard = ptgt + BT_N;
    float* rowSoft = rowHard + BT_N;
    float* rowValid = rowSoft + BT_N;

    // ---- one-shot f32 -> bf16 conversions ----
    cvt_bf16_kernel<<<(int)(nsi / 4 / 256), 256, 0, stream>>>(si, bsi, (int)(nsi / 4));
    cvt_bf16_kernel<<<(int)(nti / 4 / 256), 256, 0, stream>>>(ti, bti, (int)(nti / 4));
    cvt_bf16_kernel<<<(int)(nwS / 4 / 256), 256, 0, stream>>>(wS, bwS, (int)(nwS / 4));
    cvt_bf16_kernel<<<(int)(nwT / 4 / 256), 256, 0, stream>>>(wT, bwT, (int)(nwT / 4));

    const size_t shmem = (size_t)2 * ROWS * LDOUT * sizeof(float)
                       + (size_t)(ROWS + COLS) * LDAB * sizeof(unsigned short);

    dim3 g1(NRB, NCB);   // x = row blocks fastest -> weight slab reuse in L2
    distill_gemm_stats_kernel<<<g1, 256, shmem, stream>>>(
        bsi, bti, tg, bwS, bS, bwT, bT, pm, pse, pdot, pss, ptt, ptgt);

    row_finalize_kernel<<<BT_N / 256, 256, 0, stream>>>(
        pm, pse, pdot, pss, ptt, ptgt, tg, rowHard, rowSoft, rowValid);

    final_reduce_kernel<<<1, 256, 0, stream>>>(rowHard, rowSoft, rowValid, out);
}
